// OSQP_31044023615691
// MI455X (gfx1250) — compile-verified, matrix operand order checked by reading, not run on hardware
//
#include <hip/hip_runtime.h>
#include <hip/hip_bf16.h>

// ---------------------------------------------------------------------------
// Problem constants (match reference)
// ---------------------------------------------------------------------------
constexpr int kB  = 64;    // batch
constexpr int kN  = 256;   // variables
constexpr int kM  = 512;   // constraints
constexpr int kIters = 200;
constexpr int kNS    = 10; // Newton-Schulz iterations for M^{-1}

constexpr float kSigma = 1e-6f;
constexpr float kRho   = 0.1f;
constexpr float kAlpha = 1.6f;

typedef __bf16 bf16;
typedef __attribute__((ext_vector_type(16))) __bf16 v16bf;
typedef __attribute__((ext_vector_type(8)))  __bf16 v8bf;
typedef __attribute__((ext_vector_type(8)))  float  v8f;
typedef __attribute__((ext_vector_type(2)))  float  v2f;

// ---------------------------------------------------------------------------
// Kernel 1: convert A (f32, row-major [B, M, N]) -> bf16 copies A and A^T.
// One-time O(33 MB) pass; bf16 halves the L2 footprint of the hot loop.
// ---------------------------------------------------------------------------
__global__ void __launch_bounds__(256)
k_convert(const float* __restrict__ A, bf16* __restrict__ Abf,
          bf16* __restrict__ ATbf) {
  size_t idx = (size_t)blockIdx.x * 256 + threadIdx.x;
  const size_t total = (size_t)kB * kM * kN;
  if (idx >= total) return;
  float a = A[idx];
  bf16 v = (bf16)a;
  Abf[idx] = v;
  size_t b = idx / ((size_t)kM * kN);
  size_t r = idx - b * (size_t)kM * kN;
  size_t m = r / kN;
  size_t n = r - m * kN;
  ATbf[b * (size_t)kN * kM + n * (size_t)kM + m] = v;
}

// ---------------------------------------------------------------------------
// Kernel 2: Mmat = rho * A^T A + diag(P) + sigma*I   (per batch, 256x256 f32)
// WMMA bf16, f32 accumulate. One wave computes one 16x16 tile, K = 512.
// Both operands come from the transposed copy so every lane does two
// contiguous 16-byte loads per K-step (ISA 16-bit A-operand layout:
// lanes 0-15 hold K {0..7,16..23}, lanes 16-31 hold K {8..15,24..31}).
// ---------------------------------------------------------------------------
__global__ void __launch_bounds__(256)
k_gram(const bf16* __restrict__ ATbf, const float* __restrict__ P,
       float* __restrict__ Mm) {
  const int b    = blockIdx.y;
  const int wave = threadIdx.x >> 5;
  const int lane = threadIdx.x & 31;
  const int tile = blockIdx.x * 8 + wave;        // 0..255
  const int i0   = (tile >> 4) << 4;
  const int j0   = (tile & 15) << 4;
  const int half = lane >> 4;
  const int l16  = lane & 15;
  const int off  = half ? 8 : 0;

  const bf16* AT = ATbf + (size_t)b * kN * kM;
  float*      Mb = Mm   + (size_t)b * kN * kN;

  v8f acc = {0.f, 0.f, 0.f, 0.f, 0.f, 0.f, 0.f, 0.f};
  for (int m0 = 0; m0 < kM; m0 += 32) {
    const bf16* ra = AT + (size_t)(i0 + l16) * kM + m0 + off;
    v8bf alo = *(const v8bf*)(ra);
    v8bf ahi = *(const v8bf*)(ra + 16);
    v16bf av = __builtin_shufflevector(alo, ahi,
        0, 1, 2, 3, 4, 5, 6, 7, 8, 9, 10, 11, 12, 13, 14, 15);
    const bf16* rb = AT + (size_t)(j0 + l16) * kM + m0 + off;
    v8bf blo = *(const v8bf*)(rb);
    v8bf bhi = *(const v8bf*)(rb + 16);
    v16bf bv = __builtin_shufflevector(blo, bhi,
        0, 1, 2, 3, 4, 5, 6, 7, 8, 9, 10, 11, 12, 13, 14, 15);
    acc = __builtin_amdgcn_wmma_f32_16x16x32_bf16(
        false, av, false, bv, (short)0, acc, false, false);
  }
  const int col = j0 + l16;
#pragma unroll
  for (int v = 0; v < 8; ++v) {
    int row = i0 + v + (half ? 8 : 0);     // ISA C/D layout: v0..7 = M rows
    float val = kRho * acc[v];
    if (row == col) val += P[b * kN + row] + kSigma;
    Mb[(size_t)row * kN + col] = val;
  }
}

// ---------------------------------------------------------------------------
// Kernel 3: X0 = (1 / max_row_abs_sum(M)) * I   (Gershgorin upper bound on
// lambda_max guarantees Newton-Schulz contraction for SPD M).
// ---------------------------------------------------------------------------
__global__ void __launch_bounds__(256)
k_init(const float* __restrict__ Mm, float* __restrict__ X0) {
  const int b = blockIdx.x, t = threadIdx.x;
  const float* Mb = Mm + (size_t)b * kN * kN;
  float s = 0.f;
  for (int j = 0; j < kN; ++j)                  // column sum == row sum (sym)
    s += fabsf(Mb[(size_t)j * kN + t]);
  __shared__ float red[256];
  red[t] = s;
  __syncthreads();
  for (int w = 128; w > 0; w >>= 1) {
    if (t < w) red[t] = fmaxf(red[t], red[t + w]);
    __syncthreads();
  }
  float inv = 1.0f / red[0];
  float* Xb = X0 + (size_t)b * kN * kN;
  for (int j = 0; j < kN; ++j)
    Xb[(size_t)j * kN + t] = (j == t) ? inv : 0.f;
}

// ---------------------------------------------------------------------------
// Shared 256x256x256 f32 GEMM tile: one wave -> one 16x16 tile, K step 4
// using V_WMMA_F32_16X16X4_F32 (fp32 matrix path). ISA f32 A-operand layout:
// lanes 0-15 rows M=0..15 K={0,1}; lanes 16-31 rows M=0..15 K={2,3}.
// ---------------------------------------------------------------------------
__device__ inline v8f gemm256_f32(const float* __restrict__ Am,
                                  const float* __restrict__ Bm,
                                  int i0, int j0, int lane) {
  const int half = lane >> 4;
  const int l16  = lane & 15;
  const int rowA = i0 + l16;
  const int n    = j0 + l16;
  v8f acc = {0.f, 0.f, 0.f, 0.f, 0.f, 0.f, 0.f, 0.f};
#if __has_builtin(__builtin_amdgcn_wmma_f32_16x16x4_f32)
  for (int k0 = 0; k0 < kN; k0 += 4) {
    const int ka = k0 + (half ? 2 : 0);
    v2f av = *(const v2f*)(Am + (size_t)rowA * kN + ka);
    v2f bv;
    bv.x = Bm[(size_t)ka * kN + n];
    bv.y = Bm[(size_t)(ka + 1) * kN + n];
    acc = __builtin_amdgcn_wmma_f32_16x16x4_f32(
        false, av, false, bv, (short)0, acc, false, false);
  }
#else
  for (int k = 0; k < kN; ++k) {
    float bvv = Bm[(size_t)k * kN + n];
#pragma unroll
    for (int v = 0; v < 8; ++v)
      acc[v] += Am[(size_t)(i0 + v + (half ? 8 : 0)) * kN + k] * bvv;
  }
#endif
  return acc;
}

// Kernel 4a: Y = Mmat * X
__global__ void __launch_bounds__(256)
k_mm(const float* __restrict__ Mm, const float* __restrict__ X,
     float* __restrict__ Y) {
  const int b    = blockIdx.y;
  const int wave = threadIdx.x >> 5;
  const int lane = threadIdx.x & 31;
  const int tile = blockIdx.x * 8 + wave;
  const int i0 = (tile >> 4) << 4, j0 = (tile & 15) << 4;
  const int half = lane >> 4, l16 = lane & 15;
  const float* A = Mm + (size_t)b * kN * kN;
  const float* B = X  + (size_t)b * kN * kN;
  float*       C = Y  + (size_t)b * kN * kN;
  v8f acc = gemm256_f32(A, B, i0, j0, lane);
  const int col = j0 + l16;
#pragma unroll
  for (int v = 0; v < 8; ++v) {
    int row = i0 + v + (half ? 8 : 0);
    C[(size_t)row * kN + col] = acc[v];
  }
}

// Kernel 4b: Xn = 2*Xc - Xc*Y   (Newton-Schulz update)
__global__ void __launch_bounds__(256)
k_ns(const float* __restrict__ Xc, const float* __restrict__ Y,
     float* __restrict__ Xn) {
  const int b    = blockIdx.y;
  const int wave = threadIdx.x >> 5;
  const int lane = threadIdx.x & 31;
  const int tile = blockIdx.x * 8 + wave;
  const int i0 = (tile >> 4) << 4, j0 = (tile & 15) << 4;
  const int half = lane >> 4, l16 = lane & 15;
  const float* A = Xc + (size_t)b * kN * kN;
  const float* B = Y  + (size_t)b * kN * kN;
  float*       C = Xn + (size_t)b * kN * kN;
  v8f acc = gemm256_f32(A, B, i0, j0, lane);
  const int col = j0 + l16;
#pragma unroll
  for (int v = 0; v < 8; ++v) {
    int row = i0 + v + (half ? 8 : 0);
    C[(size_t)row * kN + col] =
        2.0f * A[(size_t)row * kN + col] - acc[v];
  }
}

// ---------------------------------------------------------------------------
// Kernel 5: 200 ADMM iterations. One workgroup (256 threads, 8 waves) per
// batch item. x,z,y live in registers; shared vectors in 4 KB LDS. All
// global GEMV reads are column-wise => one coalesced L2 line per step.
// Minv is symmetric, so reading Minv[k*kN + t] is both coalesced and the
// row needed for x_tilde[t].
// ---------------------------------------------------------------------------
__global__ void __launch_bounds__(256)
k_admm(const bf16* __restrict__ Abf, const bf16* __restrict__ ATbf,
       const float* __restrict__ Minv, const float* __restrict__ q,
       const float* __restrict__ lv, const float* __restrict__ uv,
       float* __restrict__ out) {
  const int b = blockIdx.x, t = threadIdx.x;
  const bf16*  Ab  = Abf  + (size_t)b * kM * kN;
  const bf16*  ATb = ATbf + (size_t)b * kN * kM;
  const float* Mi  = Minv + (size_t)b * kN * kN;

  const float qr = q[(size_t)b * kN + t];
  const float l0 = lv[(size_t)b * kM + t];
  const float l1 = lv[(size_t)b * kM + t + 256];
  const float u0 = uv[(size_t)b * kM + t];
  const float u1 = uv[(size_t)b * kM + t + 256];
  const float inv_rho = 1.0f / kRho;

  float xr = 0.f, z0 = 0.f, z1 = 0.f, y0 = 0.f, y1 = 0.f;

  __shared__ float sw[kM];     // rho*z - y
  __shared__ float srhs[kN];   // sigma*x - q + A^T w
  __shared__ float sxt[kN];    // x_tilde

  for (int it = 0; it < kIters; ++it) {
    sw[t]       = kRho * z0 - y0;
    sw[t + 256] = kRho * z1 - y1;
    __syncthreads();

    float acc = 0.f;
#pragma unroll 4
    for (int m = 0; m < kM; ++m)
      acc += (float)Ab[(size_t)m * kN + t] * sw[m];
    srhs[t] = kSigma * xr - qr + acc;
    __syncthreads();

    acc = 0.f;
#pragma unroll 4
    for (int k = 0; k < kN; ++k)
      acc += Mi[(size_t)k * kN + t] * srhs[k];
    sxt[t] = acc;
    __syncthreads();

    float zt0 = 0.f, zt1 = 0.f;
#pragma unroll 4
    for (int n = 0; n < kN; ++n) {
      float xv = sxt[n];
      zt0 += (float)ATb[(size_t)n * kM + t] * xv;
      zt1 += (float)ATb[(size_t)n * kM + t + 256] * xv;
    }

    xr = kAlpha * sxt[t] + (1.f - kAlpha) * xr;
    float zr0 = kAlpha * zt0 + (1.f - kAlpha) * z0;
    float zr1 = kAlpha * zt1 + (1.f - kAlpha) * z1;
    float zn0 = fminf(fmaxf(zr0 + y0 * inv_rho, l0), u0);
    float zn1 = fminf(fmaxf(zr1 + y1 * inv_rho, l1), u1);
    y0 += kRho * (zr0 - zn0);
    y1 += kRho * (zr1 - zn1);
    z0 = zn0;
    z1 = zn1;
    // next iteration's first __syncthreads() protects sw/srhs/sxt reuse
  }
  out[(size_t)b * kN + t] = xr;
}

// ---------------------------------------------------------------------------
// Host launcher: inputs are (P_val, q_val, A_val, l_val, u_val), all f32.
// Workspace layout (static offsets): Abf16 | ATbf16 | Mmat | X0 | X1 | Y
// ---------------------------------------------------------------------------
extern "C" void kernel_launch(void* const* d_in, const int* in_sizes, int n_in,
                              void* d_out, int out_size, void* d_ws,
                              size_t ws_size, hipStream_t stream) {
  (void)in_sizes; (void)n_in; (void)out_size; (void)ws_size;
  const float* P = (const float*)d_in[0];
  const float* q = (const float*)d_in[1];
  const float* A = (const float*)d_in[2];
  const float* l = (const float*)d_in[3];
  const float* u = (const float*)d_in[4];

  char* ws = (char*)d_ws;
  const size_t szBf = (size_t)kB * kM * kN * sizeof(bf16);   // 16.7 MB
  const size_t szF  = (size_t)kB * kN * kN * sizeof(float);  // 16.7 MB
  bf16*  Abf  = (bf16*)(ws);
  bf16*  ATbf = (bf16*)(ws + szBf);
  float* Mm   = (float*)(ws + 2 * szBf);
  float* X0   = (float*)(ws + 2 * szBf + szF);
  float* X1   = (float*)(ws + 2 * szBf + 2 * szF);
  float* Yb   = (float*)(ws + 2 * szBf + 3 * szF);

  const size_t total = (size_t)kB * kM * kN;
  k_convert<<<(unsigned)((total + 255) / 256), 256, 0, stream>>>(A, Abf, ATbf);
  k_gram<<<dim3(32, kB), 256, 0, stream>>>(ATbf, P, Mm);
  k_init<<<kB, 256, 0, stream>>>(Mm, X0);

  float* Xc = X0;
  float* Xn = X1;
  for (int it = 0; it < kNS; ++it) {
    k_mm<<<dim3(32, kB), 256, 0, stream>>>(Mm, Xc, Yb);
    k_ns<<<dim3(32, kB), 256, 0, stream>>>(Xc, Yb, Xn);
    float* tmp = Xc; Xc = Xn; Xn = tmp;
  }

  k_admm<<<kB, 256, 0, stream>>>(Abf, ATbf, Xc, q, l, u, (float*)d_out);
}
